// StackRNN_9835475107996
// MI455X (gfx1250) — matrix operand order repaired and checked by reading, not compiled, not run on hardware
//
#include <hip/hip_runtime.h>
#include <hip/hip_bf16.h>

// ---------------------------------------------------------------------------
// StackRNN forward for MI455X (gfx1250, wave32, WMMA).
//   - gx / logits GEMMs: v_wmma_f32_16x16x32_bf16, global_load_b128 pipelined
//   - GRU scan: persistent single-WGP WMMA kernel, h kept fp32+bf16 in LDS
//   - entire post-GRU NSTEPS loop fused into one 32-block kernel (per-batch
//     independence), stack + embedding slice resident in LDS (320KB/WGP)
// ---------------------------------------------------------------------------

typedef __attribute__((ext_vector_type(16))) __bf16 v16bf;
typedef __attribute__((ext_vector_type(8)))  float  v8f;

struct alignas(16) U4 { unsigned x, y, z, w; };
union Frag {
    U4 u[2];
    unsigned short s[16];
    v16bf v;
};

__device__ __forceinline__ unsigned short f2bf(float f) {
    unsigned u = __builtin_bit_cast(unsigned, f);
    unsigned r = u + 0x7FFFu + ((u >> 16) & 1u);   // round-to-nearest-even
    return (unsigned short)(r >> 16);
}

#define LOG2E 1.44269504088896341f

__device__ __forceinline__ float fast_sig(float x) {
    return __builtin_amdgcn_rcpf(1.f + __builtin_amdgcn_exp2f(-LOG2E * x));
}
__device__ __forceinline__ float fast_tanh(float x) {
    const float a = fabsf(x);
    const float t = __builtin_amdgcn_exp2f(-2.f * LOG2E * a);
    const float r = (1.f - t) * __builtin_amdgcn_rcpf(1.f + t);
    return copysignf(r, x);
}
__device__ __forceinline__ float fast_exp(float x) {
    return __builtin_amdgcn_exp2f(LOG2E * x);
}

// ---------------------------------------------------------------------------
// C = A * B^T (+bias): A MxK bf16 row-major, B NxK bf16 row-major, C MxN f32.
// 256 threads (8 waves); block tile 64(M) x 256(N); wave tile 32x64.
// ---------------------------------------------------------------------------
__global__ __launch_bounds__(256) void wmma_gemm_bf16(
    const unsigned short* __restrict__ A,
    const unsigned short* __restrict__ B,
    const float* __restrict__ bias,
    float* __restrict__ C,
    int M, int N, int K)
{
    const int lane = threadIdx.x & 31;
    const int wave = threadIdx.x >> 5;
    const int tl = lane & 15;
    const int hi = lane >> 4;
    const int m_base = blockIdx.y * 64 + (wave & 1) * 32;
    const int n_base = blockIdx.x * 256 + (wave >> 1) * 64;
    const int kSteps = K >> 5;

    const v8f vzero = {0.f, 0.f, 0.f, 0.f, 0.f, 0.f, 0.f, 0.f};
    v8f acc[2][4];
#pragma unroll
    for (int mi = 0; mi < 2; ++mi)
#pragma unroll
        for (int j = 0; j < 4; ++j) acc[mi][j] = vzero;

    const unsigned short* arow0 = A + (size_t)(m_base + tl) * K;
    const unsigned short* arow1 = A + (size_t)(m_base + 16 + tl) * K;

    for (int kk = 0; kk < kSteps; ++kk) {
        const int ko = kk * 32;
        Frag fa0, fa1;
        fa0.u[0] = *(const U4*)(arow0 + ko + hi * 8);
        fa0.u[1] = *(const U4*)(arow0 + ko + 16 + hi * 8);
        fa1.u[0] = *(const U4*)(arow1 + ko + hi * 8);
        fa1.u[1] = *(const U4*)(arow1 + ko + 16 + hi * 8);
#pragma unroll
        for (int j = 0; j < 4; ++j) {
            const unsigned short* brow =
                B + (size_t)(n_base + j * 16 + tl) * K + ko + hi * 8;
            Frag fb;
            fb.u[0] = *(const U4*)(brow);
            fb.u[1] = *(const U4*)(brow + 16);
            acc[0][j] = __builtin_amdgcn_wmma_f32_16x16x32_bf16(
                false, fa0.v, false, fb.v, (short)0, acc[0][j], false, false);
            acc[1][j] = __builtin_amdgcn_wmma_f32_16x16x32_bf16(
                false, fa1.v, false, fb.v, (short)0, acc[1][j], false, false);
        }
    }

#pragma unroll
    for (int j = 0; j < 4; ++j) {
        const int n = n_base + j * 16 + tl;
        const float bv = bias ? bias[n] : 0.f;
#pragma unroll
        for (int mi = 0; mi < 2; ++mi) {
#pragma unroll
            for (int i = 0; i < 8; ++i) {
                const int m = m_base + mi * 16 + hi * 8 + i;
                __builtin_nontemporal_store(acc[mi][j][i] + bv,
                                            &C[(size_t)m * N + n]);
            }
        }
    }
}

// ---------------------------------------------------------------------------
__global__ void cvt_bf16(const float* __restrict__ in,
                         unsigned short* __restrict__ out, int n)
{
    int i = blockIdx.x * blockDim.x + threadIdx.x;
    if (i < n) out[i] = f2bf(in[i]);
}

__global__ void gather_embs(const int* __restrict__ inputs,
                            const float* __restrict__ emb,
                            float* __restrict__ embs_f,
                            unsigned short* __restrict__ embs_bf)
{
    const int r = blockIdx.x;        // r = s*32 + b
    const int e = threadIdx.x;       // 0..255
    const int idx = inputs[r];
    const float v = emb[(size_t)idx * 256 + e];
    embs_f[(size_t)r * 256 + e] = v;
    embs_bf[(size_t)r * 256 + e] = f2bf(v);
}

// ---------------------------------------------------------------------------
// GRU scan: one workgroup (1024 threads = 32 waves), 128 serial steps.
// gh = h @ w_hh^T + b_hh via WMMA (96 tiles, 3 per wave), staged in LDS,
// fused fast-math gates update h (fp32 master + bf16 mirror in LDS).
// Dynamic LDS: h 32KB + gh 96KB + h_bf16 16KB = 144KB.
// ---------------------------------------------------------------------------
__global__ __launch_bounds__(1024) void gru_scan(
    const float* __restrict__ gx,              // (S*B) x 768
    const unsigned short* __restrict__ w_hh,   // 768 x 256 bf16
    const float* __restrict__ b_hh,            // 768
    float* __restrict__ buf_outs,              // (S*B) x 256 f32
    unsigned short* __restrict__ buf_outs_bf)  // (S*B) x 256 bf16
{
    extern __shared__ float smem[];
    float* h_s  = smem;                                   // 32*256
    float* gh_s = smem + 32 * 256;                        // 32*768
    unsigned short* h_bf = (unsigned short*)(smem + 32 * 256 + 32 * 768);

    const int tid = threadIdx.x;
    for (int i = tid; i < 32 * 256; i += 1024) { h_s[i] = 0.f; h_bf[i] = 0; }
    __syncthreads();

    const int lane = tid & 31;
    const int wave = tid >> 5;
    const int tl = lane & 15, hi = lane >> 4;
    const int mi = wave & 1;              // M tile (rows b 0..15 / 16..31)
    const int nbase = (wave >> 1) * 3;    // 3 N tiles per wave -> 48 tiles
    const v8f vzero = {0.f, 0.f, 0.f, 0.f, 0.f, 0.f, 0.f, 0.f};

    for (int s = 0; s < 128; ++s) {
        v8f acc[3] = {vzero, vzero, vzero};
        const unsigned short* hrow = h_bf + (mi * 16 + tl) * 256;
        for (int kk = 0; kk < 8; ++kk) {
            const int ko = kk * 32;
            Frag fa;
            fa.u[0] = *(const U4*)(hrow + ko + hi * 8);
            fa.u[1] = *(const U4*)(hrow + ko + 16 + hi * 8);
#pragma unroll
            for (int j = 0; j < 3; ++j) {
                const int n = (nbase + j) * 16 + tl;
                const unsigned short* brow = w_hh + (size_t)n * 256 + ko + hi * 8;
                Frag fb;
                fb.u[0] = *(const U4*)(brow);
                fb.u[1] = *(const U4*)(brow + 16);
                acc[j] = __builtin_amdgcn_wmma_f32_16x16x32_bf16(
                    false, fa.v, false, fb.v, (short)0, acc[j], false, false);
            }
        }
#pragma unroll
        for (int j = 0; j < 3; ++j) {
            const int n = (nbase + j) * 16 + tl;
            const float bb = b_hh[n];
#pragma unroll
            for (int i = 0; i < 8; ++i)
                gh_s[(mi * 16 + hi * 8 + i) * 768 + n] = acc[j][i] + bb;
        }
        __syncthreads();

        const float* gxs = gx + (size_t)s * 32 * 768;
        // prefetch next step's gx block (one line per thread covers 96KB)
        if (s < 127 && tid < 768)
            __builtin_prefetch(gxs + 32 * 768 + tid * 32, 0, 1);
        for (int e = tid; e < 32 * 256; e += 1024) {
            const int b = e >> 8, j = e & 255;
            const float hr = gh_s[b * 768 + j];
            const float hz = gh_s[b * 768 + 256 + j];
            const float hn = gh_s[b * 768 + 512 + j];
            const float xr = gxs[b * 768 + j];
            const float xz = gxs[b * 768 + 256 + j];
            const float xn = gxs[b * 768 + 512 + j];
            const float r = fast_sig(xr + hr);
            const float z = fast_sig(xz + hz);
            const float nn = fast_tanh(xn + r * hn);
            const float h = (1.f - z) * nn + z * h_s[b * 256 + j];
            h_s[b * 256 + j] = h;
            h_bf[b * 256 + j] = f2bf(h);
            const size_t o = (size_t)(s * 32 + b) * 256 + j;
            buf_outs[o] = h;
            buf_outs_bf[o] = f2bf(h);
        }
        __syncthreads();
    }
}

// ---------------------------------------------------------------------------
// Fused post-GRU pipeline: one block per batch element b (32 blocks).
// Does: buf_atten, to_push init, 16x {config -> chid/action -> attention ->
// stack update}, then logp[b] = logsigmoid(stack[b,0,:].top_W + top_b).
// All state (stack 64KB, embs slice 128KB w/ stride-257 padding, scratch)
// lives in LDS (~201KB of the 320KB WGP budget).
// ---------------------------------------------------------------------------
__global__ __launch_bounds__(256) void stack_loop(
    const float* __restrict__ embs_f,     // (S*B) x 256
    const float* __restrict__ buf_outs,   // (S*B) x 256
    const float* __restrict__ chid_W, const float* __restrict__ chid_b,
    const float* __restrict__ act_W,  const float* __restrict__ act_b,
    const float* __restrict__ top_W,  const float* __restrict__ top_b,
    float* __restrict__ logp)             // 32
{
    extern __shared__ float sm[];
    float* stack_s = sm;                       // 64*256 = 16384
    float* embsl   = stack_s + 64 * 256;       // 128*257 = 32896
    float* cfg     = embsl + 128 * 257;        // 768
    float* chid_s  = cfg + 768;                // 256
    float* topush  = chid_s + 256;             // 256
    float* batt    = topush + 256;             // 256
    float* p       = batt + 256;               // 80 (66 used)
    float* sc      = p + 80;                   // 128
    float* wexp    = sc + 128;                 // 128
    float* red     = wexp + 128;               // 256

    const int b = blockIdx.x, t = threadIdx.x;

    // preload this batch's embedding slice; zero stack; init to_push=embs[0,b]
    for (int s2 = 0; s2 < 128; ++s2)
        embsl[s2 * 257 + t] = embs_f[(size_t)(s2 * 32 + b) * 256 + t];
    for (int i = t; i < 64 * 256; i += 256) stack_s[i] = 0.f;
    topush[t] = embsl[t];                      // row 0, same-thread write
    __syncthreads();

    // ---- buf_atten = attention(embs[0,b], buf_outs[:,b,:]) ----
    if (t < 128) {
        const float* mr = buf_outs + (size_t)(t * 32 + b) * 256;
        float a = 0.f;
        for (int e = 0; e < 256; ++e) a += mr[e] * topush[e];
        sc[t] = a;
    }
    __syncthreads();
    {
        float mx = sc[0];
        for (int i = 1; i < 128; ++i) mx = fmaxf(mx, sc[i]);
        if (t < 128) wexp[t] = fast_exp(sc[t] - mx);
    }
    __syncthreads();
    {
        float sum = 0.f;
        for (int i = 0; i < 128; ++i) sum += wexp[i];
        const float inv = 1.f / sum;
        float acc = 0.f;
        for (int s2 = 0; s2 < 128; ++s2)
            acc += wexp[s2] * buf_outs[(size_t)(s2 * 32 + b) * 256 + t];
        batt[t] = acc * inv;
    }
    __syncthreads();

    // ---- 16 parser steps ----
    for (int it = 0; it < 16; ++it) {
        cfg[t]       = topush[t];
        cfg[256 + t] = stack_s[t];             // stack[:,0,:]
        cfg[512 + t] = batt[t];
        __syncthreads();

        {   // chid = cfg @ chid_W^T + chid_b ; action = cfg @ act_W^T + act_b
            const float* wr = chid_W + (size_t)t * 768;
            float acc = 0.f;
            for (int k = 0; k < 768; ++k) acc += cfg[k] * wr[k];
            chid_s[t] = acc + chid_b[t];
        }
        if (t < 66) {
            const float* ar = act_W + (size_t)t * 768;
            float a2 = 0.f;
            for (int k = 0; k < 768; ++k) a2 += cfg[k] * ar[k];
            p[t] = a2 + act_b[t];
        }
        __syncthreads();

        // to_push = attention(chid, embs[:,b,:])  (all-LDS)
        if (t < 128) {
            const float* er = embsl + t * 257;
            float a = 0.f;
            for (int e = 0; e < 256; ++e) a += er[e] * chid_s[e];
            sc[t] = a;
        }
        __syncthreads();
        {
            float mx = sc[0];
            for (int i = 1; i < 128; ++i) mx = fmaxf(mx, sc[i]);
            if (t < 128) wexp[t] = fast_exp(sc[t] - mx);
        }
        __syncthreads();
        {
            float sum = 0.f;
            for (int i = 0; i < 128; ++i) sum += wexp[i];
            const float inv = 1.f / sum;
            float acc = 0.f;
            for (int s2 = 0; s2 < 128; ++s2) acc += wexp[s2] * embsl[s2 * 257 + t];
            topush[t] = acc * inv;
        }
        __syncthreads();

        // stack update: thread t owns hidden channel h = t
        float st[64];
#pragma unroll
        for (int l = 0; l < 64; ++l) st[l] = stack_s[l * 256 + t];
        const float p_push = p[0], p_pass = p[65];

        float ps = 0.f, pop0 = 0.f;
        for (int n = 0; n < 64; ++n) {
            ps += st[n];
            float tpb = ps / (float)(n + 1);
            tpb = tpb > 0.f ? tpb : 0.f;
            pop0 += p[1 + n] * tpb;
        }
        stack_s[t] = pop0 + p_push * topush[t] + p_pass * st[0];
        for (int l = 1; l < 64; ++l) {
            float acc = 0.f;
            for (int n = 0; n + l < 64; ++n) acc += p[1 + n] * st[l + n];
            stack_s[l * 256 + t] = acc + p_push * st[l - 1] + p_pass * st[l];
        }
        __syncthreads();
    }

    // ---- logp[b] = logsigmoid(stack[b,0,:] . top_W + top_b) ----
    red[t] = stack_s[t] * top_W[t];
    __syncthreads();
    for (int off2 = 128; off2 > 0; off2 >>= 1) {
        if (t < off2) red[t] += red[t + off2];
        __syncthreads();
    }
    if (t == 0) {
        const float x = red[0] + top_b[0];
        logp[b] = fminf(x, 0.f) - log1pf(expf(-fabsf(x)));
    }
}

// ---------------------------------------------------------------------------
__global__ void neg_mean_logp(const float* __restrict__ logp,
                              float* __restrict__ out_scalar)
{
    __shared__ float r[32];
    r[threadIdx.x] = logp[threadIdx.x];
    __syncthreads();
    if (threadIdx.x == 0) {
        float s = 0.f;
        for (int i = 0; i < 32; ++i) s += r[i];
        out_scalar[0] = -(s / 32.f);
    }
}

// ---------------------------------------------------------------------------
extern "C" void kernel_launch(void* const* d_in, const int* in_sizes, int n_in,
                              void* d_out, int out_size, void* d_ws, size_t ws_size,
                              hipStream_t stream)
{
    (void)in_sizes; (void)n_in; (void)out_size; (void)ws_size;
    const int*   inputs = (const int*)  d_in[0];
    const float* emb    = (const float*)d_in[1];
    const float* w_ih   = (const float*)d_in[2];
    const float* w_hh   = (const float*)d_in[3];
    const float* b_ih   = (const float*)d_in[4];
    const float* b_hh   = (const float*)d_in[5];
    const float* chid_W = (const float*)d_in[6];
    const float* chid_b = (const float*)d_in[7];
    const float* act_W  = (const float*)d_in[8];
    const float* act_b  = (const float*)d_in[9];
    const float* top_W  = (const float*)d_in[10];
    const float* top_b  = (const float*)d_in[11];
    float* out = (float*)d_out;

    size_t off = 0;
    auto carve = [&](size_t bytes) -> char* {
        char* ptr = (char*)d_ws + off;
        off += (bytes + 255) & ~(size_t)255;
        return ptr;
    };
    unsigned short* emb_bf  = (unsigned short*)carve((size_t)32000 * 256 * 2);
    unsigned short* wih_bf  = (unsigned short*)carve((size_t)768 * 256 * 2);
    unsigned short* whh_bf  = (unsigned short*)carve((size_t)768 * 256 * 2);
    float*          embs_f  = (float*)         carve((size_t)4096 * 256 * 4);
    unsigned short* embs_bf = (unsigned short*)carve((size_t)4096 * 256 * 2);
    float*          gx      = (float*)         carve((size_t)4096 * 768 * 4);
    float*          buf_f   = (float*)         carve((size_t)4096 * 256 * 4);
    unsigned short* buf_bf  = (unsigned short*)carve((size_t)4096 * 256 * 2);
    float*          logp    = (float*)         carve((size_t)32 * 4);

    cvt_bf16<<<(32000 * 256 + 255) / 256, 256, 0, stream>>>(emb, emb_bf, 32000 * 256);
    cvt_bf16<<<(768 * 256 + 255) / 256, 256, 0, stream>>>(w_ih, wih_bf, 768 * 256);
    cvt_bf16<<<(768 * 256 + 255) / 256, 256, 0, stream>>>(w_hh, whh_bf, 768 * 256);

    gather_embs<<<4096, 256, 0, stream>>>(inputs, emb, embs_f, embs_bf);

    // gx = embs @ w_ih^T + b_ih   (M=4096, N=768, K=256)
    wmma_gemm_bf16<<<dim3(768 / 256, 4096 / 64), 256, 0, stream>>>(
        embs_bf, wih_bf, b_ih, gx, 4096, 768, 256);

    // GRU scan (single WGP, 144KB dynamic LDS)
    gru_scan<<<1, 1024, 144 * 1024, stream>>>(gx, whh_bf, b_hh, buf_f, buf_bf);

    // fused attention + 16-step parser + logp  (201KB dynamic LDS per block)
    const size_t sl_lds = (size_t)(64 * 256 + 128 * 257 + 768 + 256 + 256 + 256 +
                                   80 + 128 + 128 + 256) * sizeof(float);
    stack_loop<<<32, 256, sl_lds, stream>>>(embs_f, buf_f,
                                            chid_W, chid_b, act_W, act_b,
                                            top_W, top_b, logp);

    neg_mean_logp<<<1, 32, 0, stream>>>(logp, out + (size_t)4096 * 32000);

    // logits = buf_outs @ emb^T   (M=4096, N=32000, K=256) -> d_out
    wmma_gemm_bf16<<<dim3(32000 / 256, 4096 / 64), 256, 0, stream>>>(
        buf_bf, emb_bf, nullptr, out, 4096, 32000, 256);
}
